// RVQVAE_42262478192888
// MI455X (gfx1250) — matrix operand order repaired and checked by reading, not compiled, not run on hardware
//
#include <hip/hip_runtime.h>
#include <hip/hip_bf16.h>

typedef __attribute__((ext_vector_type(16))) __bf16 v16bf;
typedef __attribute__((ext_vector_type(8)))  float  v8f;

// ---------------------------------------------------------------------------
// small utility kernels
// ---------------------------------------------------------------------------
__global__ void k_zero_f32(float* p, long n) {
  long i = (long)blockIdx.x * blockDim.x + threadIdx.x;
  if (i < n) p[i] = 0.f;
}

__global__ void k_cast_f32_bf16(const float* __restrict__ s, __bf16* __restrict__ d, long n) {
  long i = (long)blockIdx.x * blockDim.x + threadIdx.x;
  if (i < n) d[i] = (__bf16)s[i];
}

// fold BN (eval) into y = a*x + c
__global__ void k_bn_coef(const float* g, const float* be, const float* m, const float* v,
                          float* a, float* c, int C) {
  int i = blockIdx.x * blockDim.x + threadIdx.x;
  if (i < C) {
    float av = g[i] * rsqrtf(v[i] + 1e-5f);
    a[i] = av;
    c[i] = be[i] - m[i] * av;
  }
}

// ConvTranspose weight (I,O,3) -> equivalent conv weight A[o][i*3+k] = w[i][o][2-k], bf16
__global__ void k_prep_decw(const float* __restrict__ w, __bf16* __restrict__ A, int I, int O) {
  int idx = blockIdx.x * blockDim.x + threadIdx.x;
  int tot = O * I * 3;
  if (idx >= tot) return;
  int o = idx / (I * 3);
  int r = idx - o * (I * 3);
  int i = r / 3;
  int k = r - i * 3;
  A[idx] = (__bf16)w[(long)i * O * 3 + o * 3 + (2 - k)];
}

// codebooks (Q,D,K) -> eT (Q,K,D) bf16 + e2[q*K+c] = sum_i e^2
__global__ void k_prep_codebook(const float* __restrict__ cb, __bf16* __restrict__ eT,
                                float* __restrict__ e2, int NQ, int D, int K) {
  int idx = blockIdx.x * blockDim.x + threadIdx.x;
  if (idx >= NQ * K) return;
  int q = idx / K, c = idx - q * K;
  float s = 0.f;
  for (int i = 0; i < D; ++i) {
    float v = cb[(long)q * D * K + (long)i * K + c];
    eT[(long)q * K * D + (long)c * D + i] = (__bf16)v;
    s += v * v;
  }
  e2[idx] = s;
}

// ---------------------------------------------------------------------------
// unified implicit-GEMM kernel (conv1d / conv_transpose1d / VQ distance GEMM)
//   C[m, p] = sum_{gk} A[m, gk] * im2col(X)[gk, p],  f32 accumulate via WMMA bf16
// 128x128x32 block, 8 waves, each wave = 32x64 (2x4 WMMA tiles)
// ---------------------------------------------------------------------------
struct GArgs {
  const __bf16* A;      // (M x Ktot) row-major bf16 weights
  const __bf16* X;      // input activations (bf16)
  float*  outF;         // f32 output (modes 1,2,3)
  __bf16* outH;         // bf16 output (modes 0,1)
  const float* bnA;     // per-channel scale (null => skip BN)
  const float* bnC;     // per-channel shift
  int M, Ktot;
  int noShift;          // No = 1 << noShift (all No are powers of two)
  int Ni;
  int xB, xC, xP;       // input element strides
  int oB, oC;           // output element strides (channel-major modes 0/3)
  int Ntot;             // Bn << noShift
  int relu;
};

#define BM 128
#define BN 128
#define BK 32

template <int KW, int MODE, int STRIDE, int EMODE>
__global__ __launch_bounds__(256) void k_gemm_conv(GArgs p) {
  constexpr int PADLO = (KW == 3) ? 1 : 0;
  __shared__ __bf16 As[BM][BK];   // [m][k]
  __shared__ __bf16 Bs[BN][BK];   // [n][k]

  const int tid  = threadIdx.x;
  const int lane = tid & 31;
  const int wave = tid >> 5;
  const int wm = wave & 3;        // 4 M-slices of 32
  const int wn = wave >> 2;       // 2 N-slices of 64
  const int m0 = blockIdx.x * BM;
  const int n0 = blockIdx.y * BN;
  const int noMask = (1 << p.noShift) - 1;

  v8f acc[2][4];
  #pragma unroll
  for (int a = 0; a < 2; ++a)
    #pragma unroll
    for (int b = 0; b < 4; ++b) acc[a][b] = (v8f){};

  const int nK = (p.Ktot + BK - 1) / BK;
  for (int kt = 0; kt < nK; ++kt) {
    const int k0 = kt * BK;
    if (kt + 1 < nK)  // next weight tile -> global_prefetch_b8
      __builtin_prefetch(p.A + (long)m0 * p.Ktot + (k0 + BK), 0, 1);

    // ---- stage A (weights): 128x32, 16/thread, branchless
    #pragma unroll
    for (int it = 0; it < (BM * BK) / 256; ++it) {
      int i  = tid + it * 256;
      int mr = i >> 5, kk = i & 31;
      int gm = m0 + mr, gk = k0 + kk;
      bool ok = (gm < p.M) && (gk < p.Ktot);
      int gmc = ok ? gm : 0;
      int gkc = ok ? gk : 0;
      float v = (float)p.A[(long)gmc * p.Ktot + gkc];
      As[mr][kk] = (__bf16)(ok ? v : 0.f);
    }
    // ---- stage B (implicit im2col): 128x32, branchless (clamped addr + select)
    #pragma unroll
    for (int it = 0; it < (BN * BK) / 256; ++it) {
      int i  = tid + it * 256;
      int nr = i >> 5, kk = i & 31;
      int pg = n0 + nr;
      int gk = k0 + kk;
      bool ok = (pg < p.Ntot) && (gk < p.Ktot);
      int pgc = (pg < p.Ntot) ? pg : (p.Ntot - 1);
      int gkc = (gk < p.Ktot) ? gk : (p.Ktot - 1);
      int ci, kw;
      if (KW == 1) { ci = gkc; kw = 0; }
      else { ci = gkc / 3; kw = gkc - ci * 3; }
      int b   = pgc >> p.noShift;
      int pos = pgc & noMask;
      int src;
      if (MODE == 0) {                      // conv
        src = pos * STRIDE + kw - PADLO;
        ok = ok && (src >= 0) && (src < p.Ni);
      } else {                              // convT over dilated input
        int t = pos + kw - PADLO;
        ok = ok && (t >= 0);
        if (STRIDE == 2) { ok = ok && ((t & 1) == 0); src = t >> 1; }
        else src = t;
        ok = ok && (src < p.Ni);
      }
      int srcc = (src >= 0 && src < p.Ni) ? src : 0;
      float v = (float)p.X[(long)b * p.xB + ci * p.xC + srcc * p.xP];
      Bs[nr][kk] = (__bf16)(ok ? v : 0.f);
    }
    __syncthreads();

    // ---- pack WMMA operands per CDNA5 16-bit layouts
    v16bf av[2], bv[4];
    {
      const int mrow0 = wm * 32 + (lane & 15);
      const int kau   = (lane >= 16) ? 8 : 0;   // A: lanes 16-31 hold K+8 / K+24
      const int ncol0 = wn * 64 + (lane & 15);
      const int kbu   = (lane >= 16) ? 8 : 0;   // B: lanes 16-31 hold K+8 / K+24
      #pragma unroll
      for (int e = 0; e < 16; ++e) {
        int vg = e >> 1, lo = e & 1;
        int ka = ((vg & 3) * 2 + lo) + ((vg & 4) ? 16 : 0) + kau;
        #pragma unroll
        for (int a = 0; a < 2; ++a) av[a][e] = As[mrow0 + a * 16][ka];
        #pragma unroll
        for (int b = 0; b < 4; ++b) bv[b][e] = Bs[ncol0 + b * 16][e + kbu + ((e >= 8) ? 8 : 0)];
      }
    }
    #pragma unroll
    for (int a = 0; a < 2; ++a)
      #pragma unroll
      for (int b = 0; b < 4; ++b)
        acc[a][b] = __builtin_amdgcn_wmma_f32_16x16x32_bf16(
            false, av[a], false, bv[b], (short)0, acc[a][b], false, false);
    __syncthreads();
  }

  // ---- fused epilogue (bias is zero in this model; ReLU then BN per reference)
  const int rbase = (lane >= 16) ? 8 : 0;
  const int cl = lane & 15;
  #pragma unroll
  for (int a = 0; a < 2; ++a) {
    #pragma unroll
    for (int b = 0; b < 4; ++b) {
      #pragma unroll
      for (int r = 0; r < 8; ++r) {
        int m  = m0 + wm * 32 + a * 16 + rbase + r;
        int pg = n0 + wn * 64 + b * 16 + cl;
        if (m >= p.M || pg >= p.Ntot) continue;
        float v = acc[a][b][r];
        if (p.relu) v = fmaxf(v, 0.f);
        if (p.bnA)  v = v * p.bnA[m] + p.bnC[m];
        int bb  = pg >> p.noShift;
        int pos = pg & noMask;
        if (EMODE == 0) {
          p.outH[(long)bb * p.oB + m * p.oC + pos] = (__bf16)v;
        } else if (EMODE == 1) {
          p.outF[(long)pg * p.M + m] = v;
          p.outH[(long)pg * p.M + m] = (__bf16)v;
        } else if (EMODE == 2) {
          p.outF[(long)pg * p.M + m] = v;
        } else {
          p.outF[(long)bb * p.oB + m * p.oC + pos] = v;
        }
      }
    }
  }
}

// ---------------------------------------------------------------------------
// VQ: per row j, argmin_c (||e_c||^2 - 2*G[j,c]); fused token emit +
// residual/q_sum update + commit-loss reduction (one wave per row)
// ---------------------------------------------------------------------------
__global__ __launch_bounds__(256) void k_vq_update(
    const float* __restrict__ G, const float* __restrict__ e2,
    const __bf16* __restrict__ eT,
    float* resF, __bf16* resH, float* qsF,
    float* tokOut, float* commitOut,
    int D, int KEc, int rows, float commitScale) {
  int row  = (int)((((long)blockIdx.x * blockDim.x) + threadIdx.x) >> 5);
  int lane = threadIdx.x & 31;
  if (row >= rows) return;

  float best = 3.4e38f; int bidx = 0;
  const float* g = G + (long)row * KEc;
  for (int c = lane; c < KEc; c += 32) {
    float d = e2[c] - 2.f * g[c];
    if (d < best) { best = d; bidx = c; }
  }
  #pragma unroll
  for (int off = 16; off >= 1; off >>= 1) {
    float ov = __shfl_xor(best, off, 32);
    int   oi = __shfl_xor(bidx, off, 32);
    if (ov < best || (ov == best && oi < bidx)) { best = ov; bidx = oi; }
  }

  const __bf16* ev = eT + (long)bidx * D;
  float csum = 0.f;
  for (int i = lane; i < D; i += 32) {
    float qv = (float)ev[i];
    float r  = resF[(long)row * D + i];
    float diff = qv - r;            // quant - residual (pre-update)
    csum += diff * diff;
    resF[(long)row * D + i] = -diff;       // residual - quant
    resH[(long)row * D + i] = (__bf16)(-diff);
    qsF[(long)row * D + i] += qv;          // q_sum += quant
  }
  #pragma unroll
  for (int off = 16; off >= 1; off >>= 1) csum += __shfl_xor(csum, off, 32);
  if (lane == 0) {
    ((int*)tokOut)[row] = bidx;
    atomicAdd(commitOut, csum * commitScale);
  }
}

// ---------------------------------------------------------------------------
// host-side orchestration
// ---------------------------------------------------------------------------
static inline long cdiv(long a, long b) { return (a + b - 1) / b; }

template <int KW, int MODE, int STRIDE, int EMODE>
static void launch_gemm(hipStream_t stream, const __bf16* A, const __bf16* X,
                        float* oF, __bf16* oH, const float* ba, const float* bc,
                        int M, int Ktot, int Bn, int noShift, int Ni,
                        int xB, int xC, int xP, int oB, int oC, int relu) {
  GArgs a{A, X, oF, oH, ba, bc, M, Ktot, noShift, Ni, xB, xC, xP, oB, oC,
          Bn << noShift, relu};
  dim3 grid((unsigned)cdiv(M, BM), (unsigned)cdiv(a.Ntot, BN));
  k_gemm_conv<KW, MODE, STRIDE, EMODE><<<grid, dim3(256), 0, stream>>>(a);
}

extern "C" void kernel_launch(void* const* d_in, const int* in_sizes, int n_in,
                              void* d_out, int out_size, void* d_ws, size_t ws_size,
                              hipStream_t stream) {
  (void)in_sizes; (void)n_in; (void)out_size; (void)ws_size;
  const int Bb = 64, NP = 2048, DP = 263, HID = 512, LAT = 256, KE = 512, NQ = 6;
  const int NL = 512;                       // latent positions per batch
  const long FLATN = (long)Bb * NL;         // 32768

  const float* x = (const float*)d_in[0];
  const float *encW[4], *encG[4], *encBe[4], *encM[4], *encV[4];
  for (int i = 0; i < 4; ++i) {
    int b0 = 1 + i * 6;
    encW[i] = (const float*)d_in[b0];      encG[i] = (const float*)d_in[b0 + 2];
    encBe[i] = (const float*)d_in[b0 + 3]; encM[i] = (const float*)d_in[b0 + 4];
    encV[i] = (const float*)d_in[b0 + 5];
  }
  const float *decW[4], *decG[3], *decBe[3], *decM[3], *decV[3];
  for (int i = 0; i < 3; ++i) {
    int b0 = 25 + i * 6;
    decW[i] = (const float*)d_in[b0];      decG[i] = (const float*)d_in[b0 + 2];
    decBe[i] = (const float*)d_in[b0 + 3]; decM[i] = (const float*)d_in[b0 + 4];
    decV[i] = (const float*)d_in[b0 + 5];
  }
  decW[3] = (const float*)d_in[43];
  const float* cb = (const float*)d_in[45];

  // ---- workspace carve
  char* wsp = (char*)d_ws;
  auto alloc = [&](size_t bytes) -> char* {
    char* r = wsp; wsp += (bytes + 255) & ~(size_t)255; return r;
  };
  const int encCin[4] = {DP, HID, HID, HID}, encCout[4] = {HID, HID, HID, LAT};
  const int decCin[4] = {LAT, HID, HID, HID}, decCout[4] = {HID, HID, HID, DP};
  __bf16* wEnc[4]; __bf16* wDec[4];
  for (int i = 0; i < 4; ++i) wEnc[i] = (__bf16*)alloc((size_t)encCout[i] * encCin[i] * 3 * 2);
  for (int i = 0; i < 4; ++i) wDec[i] = (__bf16*)alloc((size_t)decCout[i] * decCin[i] * 3 * 2);
  __bf16* eT  = (__bf16*)alloc((size_t)NQ * KE * LAT * 2);
  float*  e2  = (float*)alloc((size_t)NQ * KE * 4);
  float* bnAe[4]; float* bnCe[4]; float* bnAd[3]; float* bnCd[3];
  for (int i = 0; i < 4; ++i) { bnAe[i] = (float*)alloc(HID * 4); bnCe[i] = (float*)alloc(HID * 4); }
  for (int i = 0; i < 3; ++i) { bnAd[i] = (float*)alloc(HID * 4); bnCd[i] = (float*)alloc(HID * 4); }
  const size_t actElems = (size_t)Bb * HID * NP;
  __bf16* actA = (__bf16*)alloc(actElems * 2);
  __bf16* actB = (__bf16*)alloc(actElems * 2);
  float*  resF = (float*)alloc((size_t)FLATN * LAT * 4);
  __bf16* resH = (__bf16*)alloc((size_t)FLATN * LAT * 2);
  float*  qsF  = (float*)alloc((size_t)FLATN * LAT * 4);
  __bf16* qsH  = (__bf16*)alloc((size_t)FLATN * LAT * 2);
  float*  Gbuf = (float*)alloc((size_t)FLATN * KE * 4);

  float* hOut   = (float*)d_out;                   // (64,263,2048)
  float* tokOut = hOut + (long)Bb * DP * NP;       // (6,64,512) int bits
  float* cOut   = tokOut + (long)NQ * FLATN;       // scalar

  // ---- prep: casts, BN folding, codebook transpose
  { long n = (long)Bb * DP * NP;
    k_cast_f32_bf16<<<dim3((unsigned)cdiv(n, 256)), dim3(256), 0, stream>>>(x, actA, n); }
  for (int i = 0; i < 4; ++i) {
    long n = (long)encCout[i] * encCin[i] * 3;
    k_cast_f32_bf16<<<dim3((unsigned)cdiv(n, 256)), dim3(256), 0, stream>>>(encW[i], wEnc[i], n);
    k_bn_coef<<<dim3(2), dim3(256), 0, stream>>>(encG[i], encBe[i], encM[i], encV[i],
                                                 bnAe[i], bnCe[i], encCout[i]);
  }
  for (int i = 0; i < 4; ++i) {
    long n = (long)decCout[i] * decCin[i] * 3;
    k_prep_decw<<<dim3((unsigned)cdiv(n, 256)), dim3(256), 0, stream>>>(decW[i], wDec[i],
                                                                        decCin[i], decCout[i]);
    if (i < 3)
      k_bn_coef<<<dim3(2), dim3(256), 0, stream>>>(decG[i], decBe[i], decM[i], decV[i],
                                                   bnAd[i], bnCd[i], decCout[i]);
  }
  k_prep_codebook<<<dim3((unsigned)cdiv(NQ * KE, 128)), dim3(128), 0, stream>>>(cb, eT, e2, NQ, LAT, KE);
  k_zero_f32<<<dim3((unsigned)cdiv(FLATN * (long)LAT, 256)), dim3(256), 0, stream>>>(qsF, FLATN * (long)LAT);
  k_zero_f32<<<dim3(1), dim3(32), 0, stream>>>(cOut, 1);

  // ---- encoder: conv + ReLU + BN fused, ping-pong bf16 activations
  // L0: (64,263,2048) -> (64,512,1024), stride 2
  launch_gemm<3, 0, 2, 0>(stream, wEnc[0], actA, nullptr, actB, bnAe[0], bnCe[0],
                          HID, DP * 3, Bb, 10, NP, DP * NP, NP, 1, HID * 1024, 1024, 1);
  // L1: -> (64,512,512), stride 2
  launch_gemm<3, 0, 2, 0>(stream, wEnc[1], actB, nullptr, actA, bnAe[1], bnCe[1],
                          HID, HID * 3, Bb, 9, 1024, HID * 1024, 1024, 1, HID * 512, 512, 1);
  // L2: -> (64,512,512), stride 1
  launch_gemm<3, 0, 1, 0>(stream, wEnc[2], actA, nullptr, actB, bnAe[2], bnCe[2],
                          HID, HID * 3, Bb, 9, 512, HID * 512, 512, 1, HID * 512, 512, 1);
  // L3: -> flat residual (f32 + bf16), stride 1
  launch_gemm<3, 0, 1, 1>(stream, wEnc[3], actB, resF, resH, bnAe[3], bnCe[3],
                          LAT, HID * 3, Bb, 9, 512, HID * 512, 512, 1, 0, 0, 1);

  // ---- residual VQ: 6x (WMMA GEMM G = flat @ E^T, fused argmin/update)
  const float commitScale = 1.0f / (float)((long)Bb * LAT * NL);
  for (int q = 0; q < NQ; ++q) {
    launch_gemm<1, 0, 1, 2>(stream, eT + (long)q * KE * LAT, resH, Gbuf, nullptr,
                            nullptr, nullptr, KE, LAT, 1, 15, (int)FLATN,
                            0, 1, LAT, 0, 0, 0);
    k_vq_update<<<dim3((unsigned)cdiv(FLATN * 32, 256)), dim3(256), 0, stream>>>(
        Gbuf, e2 + (long)q * KE, eT + (long)q * KE * LAT,
        resF, resH, qsF, tokOut + (long)q * FLATN, cOut,
        LAT, KE, (int)FLATN, commitScale);
  }
  { long n = FLATN * (long)LAT;
    k_cast_f32_bf16<<<dim3((unsigned)cdiv(n, 256)), dim3(256), 0, stream>>>(qsF, qsH, n); }

  // ---- decoder: convT as implicit GEMM; ReLU+BN fused on first 3 layers
  // L0: q_sum flat (pos-major) -> (64,512,1024), stride 2
  launch_gemm<3, 1, 2, 0>(stream, wDec[0], qsH, nullptr, actA, bnAd[0], bnCd[0],
                          HID, LAT * 3, Bb, 10, NL, NL * LAT, 1, LAT, HID * 1024, 1024, 1);
  // L1: -> (64,512,2048), stride 2
  launch_gemm<3, 1, 2, 0>(stream, wDec[1], actA, nullptr, actB, bnAd[1], bnCd[1],
                          HID, HID * 3, Bb, 11, 1024, HID * 1024, 1024, 1, HID * 2048, 2048, 1);
  // L2: -> (64,512,2048), stride 1
  launch_gemm<3, 1, 1, 0>(stream, wDec[2], actB, nullptr, actA, bnAd[2], bnCd[2],
                          HID, HID * 3, Bb, 11, 2048, HID * 2048, 2048, 1, HID * 2048, 2048, 1);
  // L3: -> h (64,263,2048) f32 to d_out, no relu/bn
  launch_gemm<3, 1, 1, 3>(stream, wDec[3], actA, hOut, nullptr, nullptr, nullptr,
                          DP, HID * 3, Bb, 11, 2048, HID * 2048, 2048, 1, DP * 2048, 2048, 0);
}
